// TensorNetwork_33732673143511
// MI455X (gfx1250) — 8-bit matrix operand order repaired and checked by reading, compiled, not run on hardware
//
#include <hip/hip_runtime.h>
#include <hip/hip_bf16.h>

typedef __attribute__((ext_vector_type(8))) int v8i;

#define M_CHECK 20
#define N_BIT   48
#define BATCH   8192
#define NH      (1u << M_CHECK)

__device__ __forceinline__ int nib2bytes(unsigned x) {
  return (int)((x * 0x00204081u) & 0x01010101u);
}

__global__ __launch_bounds__(256) void ldpc_weights_wmma(
    const int* __restrict__ pcm,
    const float* __restrict__ priors,
    float* __restrict__ w)
{
  __shared__ unsigned s_mask[N_BIT];
  __shared__ float s_g0[N_BIT], s_g1[N_BIT];
  const int tid = threadIdx.x;
  if (tid < N_BIT) {
    unsigned m = 0u;
    for (int i = 0; i < M_CHECK; ++i)
      m |= (unsigned)(pcm[i * N_BIT + tid] & 1) << i;
    s_mask[tid] = m;
    float p  = 1.0f / (1.0f + expf(-priors[tid]));
    float mx = fmaxf(p, 1.0f - p);
    s_g0[tid] = 1.0f / mx;
    s_g1[tid] = (1.0f - 2.0f * p) / mx;
  }
  __syncthreads();

  const int lane = tid & 31;
  const bool lo  = (lane < 16);
  const unsigned wave = blockIdx.x * (blockDim.x >> 5) + (unsigned)(tid >> 5);
  const unsigned base = wave << 4;
  const int n = lane & 15;
  const unsigned h = base + (unsigned)n;

  v8i a = {};
  a[0] = nib2bytes(lo ? (h & 15u)         : ((h >> 16) & 15u));
  a[1] = lo ? nib2bytes((h >> 4) & 15u)   : 0;
  a[2] = lo ? nib2bytes((h >> 8) & 15u)   : 0;
  a[3] = lo ? nib2bytes((h >> 12) & 15u)  : 0;

  const unsigned mk0 = s_mask[n], mk1 = s_mask[16 + n], mk2 = s_mask[32 + n];
  v8i b0 = {}, b1 = {}, b2 = {};
  b0[0] = nib2bytes(lo ? (mk0 & 15u) : ((mk0 >> 16) & 15u));
  b1[0] = nib2bytes(lo ? (mk1 & 15u) : ((mk1 >> 16) & 15u));
  b2[0] = nib2bytes(lo ? (mk2 & 15u) : ((mk2 >> 16) & 15u));
  b0[1] = lo ? nib2bytes((mk0 >> 4) & 15u)  : 0;
  b1[1] = lo ? nib2bytes((mk1 >> 4) & 15u)  : 0;
  b2[1] = lo ? nib2bytes((mk2 >> 4) & 15u)  : 0;
  b0[2] = lo ? nib2bytes((mk0 >> 8) & 15u)  : 0;
  b1[2] = lo ? nib2bytes((mk1 >> 8) & 15u)  : 0;
  b2[2] = lo ? nib2bytes((mk2 >> 8) & 15u)  : 0;
  b0[3] = lo ? nib2bytes((mk0 >> 12) & 15u) : 0;
  b1[3] = lo ? nib2bytes((mk1 >> 12) & 15u) : 0;
  b2[3] = lo ? nib2bytes((mk2 >> 12) & 15u) : 0;

  v8i cz = {};
  v8i d0 = __builtin_amdgcn_wmma_i32_16x16x64_iu8(false, a, false, b0, cz, false, false);
  v8i d1 = __builtin_amdgcn_wmma_i32_16x16x64_iu8(false, a, false, b1, cz, false, false);
  v8i d2 = __builtin_amdgcn_wmma_i32_16x16x64_iu8(false, a, false, b2, cz, false, false);

  const float g0a = s_g0[n],      g1a = s_g1[n];
  const float g0b = s_g0[16 + n], g1b = s_g1[16 + n];
  const float g0c = s_g0[32 + n], g1c = s_g1[32 + n];

  float outv = 0.0f;
  #pragma unroll
  for (int r = 0; r < 8; ++r) {
    float f0 = (d0[r] & 1) ? g1a : g0a;
    float f1 = (d1[r] & 1) ? g1b : g0b;
    float f2 = (d2[r] & 1) ? g1c : g0c;
    float prod = f0 * f1 * f2;
    prod *= __shfl_xor(prod, 1, 32);
    prod *= __shfl_xor(prod, 2, 32);
    prod *= __shfl_xor(prod, 4, 32);
    prod *= __shfl_xor(prod, 8, 32);
    if (n == r) outv = prod;
  }
  if (lane < 8)                     w[base + (unsigned)lane] = outv;
  else if (lane >= 16 && lane < 24) w[base + 8u + (unsigned)(lane - 16)] = outv;
}

__global__ __launch_bounds__(1024) void fwht_lo(float* __restrict__ w) {
  __shared__ float lds[16384];
  const unsigned base = (unsigned)blockIdx.x << 14;
  for (int k = threadIdx.x; k < 16384; k += 1024) lds[k] = w[base + (unsigned)k];
  for (int s = 0; s < 14; ++s) {
    __syncthreads();
    const int stride = 1 << s;
    for (int p = threadIdx.x; p < 8192; p += 1024) {
      int i = ((p >> s) << (s + 1)) | (p & (stride - 1));
      float x = lds[i], y = lds[i + stride];
      lds[i] = x + y;
      lds[i + stride] = x - y;
    }
  }
  __syncthreads();
  for (int k = threadIdx.x; k < 16384; k += 1024) w[base + (unsigned)k] = lds[k];
}

__global__ __launch_bounds__(256) void fwht_hi(float* __restrict__ w) {
  const unsigned l = blockIdx.x * 256u + (unsigned)threadIdx.x;
  float x[64];
  #pragma unroll
  for (int c = 0; c < 64; ++c) x[c] = w[l + ((unsigned)c << 14)];
  #pragma unroll
  for (int s = 0; s < 6; ++s) {
    const int stride = 1 << s;
    #pragma unroll
    for (int p = 0; p < 32; ++p) {
      int i = ((p >> s) << (s + 1)) | (p & (stride - 1));
      float a = x[i], b = x[i + stride];
      x[i] = a + b;
      x[i + stride] = a - b;
    }
  }
  #pragma unroll
  for (int c = 0; c < 64; ++c) w[l + ((unsigned)c << 14)] = x[c];
}

__global__ __launch_bounds__(256) void ldpc_finalize(
    const int* __restrict__ syn,
    const float* __restrict__ priors,
    const float* __restrict__ W,
    float* __restrict__ out)
{
  __shared__ float red[256];
  const int t = threadIdx.x;

  float ls = 0.0f;
  for (int j = t; j < N_BIT; j += 256) {
    float p = 1.0f / (1.0f + expf(-priors[j]));
    ls += logf(fmaxf(p, 1.0f - p));
  }
  red[t] = ls;
  __syncthreads();
  for (int off = 128; off > 0; off >>= 1) {
    if (t < off) red[t] += red[t + off];
    __syncthreads();
  }
  const float log_scale = red[0];
  __syncthreads();

  float acc = 0.0f;
  for (int b = t; b < BATCH; b += 256) {
    const int* s = syn + b * M_CHECK;
    int idx = 0;
    #pragma unroll
    for (int i = 0; i < M_CHECK; ++i) idx |= (s[i] & 1) << i;
    float val = W[idx] * (1.0f / 1048576.0f);
    acc += logf(val + 1e-30f) + log_scale;
  }
  red[t] = acc;
  __syncthreads();
  for (int off = 128; off > 0; off >>= 1) {
    if (t < off) red[t] += red[t + off];
    __syncthreads();
  }
  if (t == 0) out[0] = -red[0] / (float)BATCH;
}

extern "C" void kernel_launch(void* const* d_in, const int* in_sizes, int n_in,
                              void* d_out, int out_size, void* d_ws, size_t ws_size,
                              hipStream_t stream) {
  const int*   pcm    = (const int*)d_in[0];
  const int*   syn    = (const int*)d_in[1];
  const float* priors = (const float*)d_in[2];
  float*       out    = (float*)d_out;
  float*       w      = (float*)d_ws;

  ldpc_weights_wmma<<<NH / 128, 256, 0, stream>>>(pcm, priors, w);
  fwht_lo<<<NH / 16384, 1024, 0, stream>>>(w);
  fwht_hi<<<16384 / 256, 256, 0, stream>>>(w);
  ldpc_finalize<<<1, 256, 0, stream>>>(syn, priors, w, out);
}
